// MoEEPTop2_44504451121358
// MI455X (gfx1250) — matrix operand order repaired and checked by reading, not compiled, not run on hardware
//
#include <hip/hip_runtime.h>
#include <hip/hip_bf16.h>
#include <math.h>

// Problem constants (B,S,D,H,E) = (4,2048,512,1024,8)
#define N_TOK   8192      // B*S
#define D_DIM   512
#define H_DIM   1024
#define E_NUM   8
#define TILE_M  64        // tokens per workgroup tile (4 x 16-row WMMA sub-tiles)
#define M_SUB   4         // TILE_M / 16
#define MAX_TILES (N_TOK / TILE_M)   // 128 candidate tiles per expert

typedef __attribute__((ext_vector_type(16))) __bf16 v16bf;
typedef __attribute__((ext_vector_type(8)))  __bf16 bf16x8;
typedef __attribute__((ext_vector_type(8)))  float  v8f;

// fp32 -> bf16, round-to-nearest-even
static __device__ __forceinline__ __bf16 f2bf(float f) {
    union { float f; unsigned u; } v; v.f = f;
    unsigned r = v.u + 0x7FFFu + ((v.u >> 16) & 1u);
    unsigned short h = (unsigned short)(r >> 16);
    return __builtin_bit_cast(__bf16, h);
}

static __device__ __forceinline__ v16bf cat16(bf16x8 lo, bf16x8 hi) {
    return __builtin_shufflevector(lo, hi, 0,1,2,3,4,5,6,7,8,9,10,11,12,13,14,15);
}

// ---------------------------------------------------------------------------
// Weight conversion: W1 [E,D,H] f32 -> W1t [E,H,D] bf16  (B-operand: N-major)
// ---------------------------------------------------------------------------
__global__ __launch_bounds__(256) void convert_w1_kernel(const float* __restrict__ W1,
                                                         __bf16* __restrict__ W1t) {
    int idx = blockIdx.x * 256 + threadIdx.x;          // over E*H*D = 4194304
    int e = idx >> 19;                                 // H*D = 524288 = 2^19
    int r = idx & 524287;
    int h = r >> 9;                                    // D = 512
    int d = r & 511;
    W1t[idx] = f2bf(W1[(size_t)e * 524288 + (size_t)d * H_DIM + h]);
}

// W2 [E,H,D] f32 -> W2t [E,D,H] bf16
__global__ __launch_bounds__(256) void convert_w2_kernel(const float* __restrict__ W2,
                                                         __bf16* __restrict__ W2t) {
    int idx = blockIdx.x * 256 + threadIdx.x;          // over E*D*H
    int e = idx >> 19;
    int r = idx & 524287;
    int d = r >> 10;                                   // H = 1024
    int h = r & 1023;
    W2t[idx] = f2bf(W2[(size_t)e * 524288 + (size_t)h * D_DIM + d]);
}

// ---------------------------------------------------------------------------
// Router: one wave32 per token. logits -> softmax -> top2 -> scatter.
// ---------------------------------------------------------------------------
__global__ __launch_bounds__(256) void router_kernel(const float* __restrict__ x,
                                                     const float* __restrict__ Wr,
                                                     const float* __restrict__ br,
                                                     int* __restrict__ cnt,
                                                     int* __restrict__ tok_list,
                                                     float* __restrict__ gate_list) {
    __shared__ float wr[E_NUM * D_DIM];   // 16 KB, whole router weight
    __shared__ float brs[E_NUM];
    int tid = threadIdx.x;
    for (int i = tid; i < E_NUM * D_DIM; i += 256) wr[i] = Wr[i];
    if (tid < E_NUM) brs[tid] = br[tid];
    __syncthreads();

    int wave = tid >> 5;
    int lane = tid & 31;
    int t = blockIdx.x * 8 + wave;

    float acc[E_NUM];
#pragma unroll
    for (int e = 0; e < E_NUM; ++e) acc[e] = 0.0f;
    const float* xr = x + (size_t)t * D_DIM;
    for (int k = lane; k < D_DIM; k += 32) {
        float xv = xr[k];
#pragma unroll
        for (int e = 0; e < E_NUM; ++e) acc[e] = fmaf(xv, wr[e * D_DIM + k], acc[e]);
    }
#pragma unroll
    for (int off = 16; off > 0; off >>= 1) {
#pragma unroll
        for (int e = 0; e < E_NUM; ++e) acc[e] += __shfl_xor(acc[e], off, 32);
    }

    if (lane == 0) {
        float l[E_NUM], m = -3.0e38f;
#pragma unroll
        for (int e = 0; e < E_NUM; ++e) { l[e] = acc[e] + brs[e]; m = fmaxf(m, l[e]); }
        float s = 0.0f;
#pragma unroll
        for (int e = 0; e < E_NUM; ++e) { l[e] = __expf(l[e] - m); s += l[e]; }
        float inv = 1.0f / s;
#pragma unroll
        for (int e = 0; e < E_NUM; ++e) l[e] *= inv;
        // top-2, first-occurrence tie-break (torch.topk semantics)
        int i1 = 0; float p1 = l[0];
#pragma unroll
        for (int e = 1; e < E_NUM; ++e) if (l[e] > p1) { p1 = l[e]; i1 = e; }
        int i2 = (i1 == 0) ? 1 : 0; float p2 = l[i2];
#pragma unroll
        for (int e = 0; e < E_NUM; ++e)
            if (e != i1 && l[e] > p2) { p2 = l[e]; i2 = e; }

        int pos1 = atomicAdd(&cnt[i1], 1);
        tok_list[i1 * N_TOK + pos1] = t;
        gate_list[i1 * N_TOK + pos1] = p1;
        int pos2 = atomicAdd(&cnt[i2], 1);
        tok_list[i2 * N_TOK + pos2] = t;
        gate_list[i2 * N_TOK + pos2] = p2;
    }
}

// ---------------------------------------------------------------------------
// Grouped expert FFN: 64-token tile, 8 waves, bf16 WMMA with f32 accumulate.
// Each B fetch (weights) is reused across 4 A sub-tiles -> 4x less L2 traffic.
// ---------------------------------------------------------------------------
#define XLDS_STRIDE 520    // 512 + 8 bf16 pad (conflict-free ds_load_b128)
#define HLDS_STRIDE 1032   // 1024 + 8 bf16 pad

__global__ __launch_bounds__(256, 1) void ffn_kernel(const float* __restrict__ x,
                                                     const __bf16* __restrict__ W1t,
                                                     const float* __restrict__ b1,
                                                     const __bf16* __restrict__ W2t,
                                                     const float* __restrict__ b2,
                                                     const int* __restrict__ cnt,
                                                     const int* __restrict__ tok_list,
                                                     const float* __restrict__ gate_list,
                                                     float* __restrict__ y) {
    int e    = blockIdx.x / MAX_TILES;
    int tile = blockIdx.x % MAX_TILES;
    int count = cnt[e];
    if (tile * TILE_M >= count) return;   // uniform per block: EXEC stays all-1s

    __shared__ __bf16 xlds[TILE_M * XLDS_STRIDE];   // ~65 KB
    __shared__ __bf16 hlds[TILE_M * HLDS_STRIDE];   // ~129 KB  (320 KB LDS/WGP)
    __shared__ int    tokLds[TILE_M];
    __shared__ float  gateLds[TILE_M];

    int tid = threadIdx.x;
    if (tid < TILE_M) {
        int ti = tile * TILE_M + tid;
        bool valid = ti < count;
        tokLds[tid]  = valid ? tok_list[e * N_TOK + ti]  : 0;
        gateLds[tid] = valid ? gate_list[e * N_TOK + ti] : 0.0f;  // gate 0 kills pad rows
    }
    __syncthreads();

    // Stage x tile (gather rows, fp32 -> bf16), coalesced over columns.
    for (int i = tid; i < TILE_M * D_DIM; i += 256) {
        int row = i >> 9, col = i & 511;
        float v = x[(size_t)tokLds[row] * D_DIM + col];
        xlds[row * XLDS_STRIDE + col] = f2bf(v);
    }
    __syncthreads();

    int wave   = tid >> 5;
    int lane   = tid & 31;
    int laneN  = lane & 15;
    int hiHalf = lane >> 4;
    int rowA   = laneN;            // A-matrix: row M = lane%16 within sub-tile
    int kbA    = hiHalf * 8;       // A: lanes16-31 take K+8 / K+24 chunks
    int kbB    = hiHalf * 8;       // B: lanes16-31 take K+8 / K+24 chunks

    // ---- FC1: h[64 x 1024] = gelu(x @ W1 + b1); wave owns 128 H-columns ----
#pragma unroll 1
    for (int nt = 0; nt < 8; ++nt) {
        int col = wave * 128 + nt * 16 + laneN;
        v8f acc[M_SUB];
#pragma unroll
        for (int mt = 0; mt < M_SUB; ++mt) acc[mt] = (v8f){};
        const __bf16* wp = W1t + ((size_t)e * H_DIM + col) * D_DIM;
#pragma unroll 2
        for (int ks = 0; ks < 16; ++ks) {
            int k0 = ks * 32;
            bf16x8 blo = *reinterpret_cast<const bf16x8*>(wp + k0 + kbB);
            bf16x8 bhi = *reinterpret_cast<const bf16x8*>(wp + k0 + kbB + 16);
            __builtin_prefetch(wp + k0 + 256, 0, 1);   // global_prefetch_b8
            v16bf b = cat16(blo, bhi);
#pragma unroll
            for (int mt = 0; mt < M_SUB; ++mt) {
                const __bf16* ap = xlds + (mt * 16 + rowA) * XLDS_STRIDE + k0 + kbA;
                bf16x8 alo = *reinterpret_cast<const bf16x8*>(ap);
                bf16x8 ahi = *reinterpret_cast<const bf16x8*>(ap + 16);
                acc[mt] = __builtin_amdgcn_wmma_f32_16x16x32_bf16(
                    false, cat16(alo, ahi), false, b, (short)0, acc[mt], false, false);
            }
        }
        float bias = b1[e * H_DIM + col];
#pragma unroll
        for (int mt = 0; mt < M_SUB; ++mt) {
#pragma unroll
            for (int r = 0; r < 8; ++r) {
                float v = acc[mt][r] + bias;
                v = 0.5f * v * (1.0f + erff(v * 0.70710678118654752f));  // exact erf-GELU
                int row = mt * 16 + r + hiHalf * 8;  // C/D layout: VGPR r, hi half -> M=r+8
                hlds[row * HLDS_STRIDE + col] = f2bf(v);
            }
        }
    }
    __syncthreads();

    // ---- FC2: o[64 x 512] = h @ W2 + b2; gate-scale; atomic combine into y ----
#pragma unroll 1
    for (int nt = 0; nt < 4; ++nt) {
        int col = wave * 64 + nt * 16 + laneN;
        v8f acc[M_SUB];
#pragma unroll
        for (int mt = 0; mt < M_SUB; ++mt) acc[mt] = (v8f){};
        const __bf16* wp = W2t + ((size_t)e * D_DIM + col) * H_DIM;
#pragma unroll 2
        for (int ks = 0; ks < 32; ++ks) {
            int k0 = ks * 32;
            bf16x8 blo = *reinterpret_cast<const bf16x8*>(wp + k0 + kbB);
            bf16x8 bhi = *reinterpret_cast<const bf16x8*>(wp + k0 + kbB + 16);
            __builtin_prefetch(wp + k0 + 256, 0, 1);
            v16bf b = cat16(blo, bhi);
#pragma unroll
            for (int mt = 0; mt < M_SUB; ++mt) {
                const __bf16* ap = hlds + (mt * 16 + rowA) * HLDS_STRIDE + k0 + kbA;
                bf16x8 alo = *reinterpret_cast<const bf16x8*>(ap);
                bf16x8 ahi = *reinterpret_cast<const bf16x8*>(ap + 16);
                acc[mt] = __builtin_amdgcn_wmma_f32_16x16x32_bf16(
                    false, cat16(alo, ahi), false, b, (short)0, acc[mt], false, false);
            }
        }
        float bias = b2[e * D_DIM + col];
#pragma unroll
        for (int mt = 0; mt < M_SUB; ++mt) {
#pragma unroll
            for (int r = 0; r < 8; ++r) {
                int row = mt * 16 + r + hiHalf * 8;
                float v = (acc[mt][r] + bias) * gateLds[row];
                // two experts contribute per token -> hardware f32 atomic add
                unsafeAtomicAdd(&y[(size_t)tokLds[row] * D_DIM + col], v);
            }
        }
    }
}

// ---------------------------------------------------------------------------
// Launch
// ---------------------------------------------------------------------------
extern "C" void kernel_launch(void* const* d_in, const int* in_sizes, int n_in,
                              void* d_out, int out_size, void* d_ws, size_t ws_size,
                              hipStream_t stream) {
    (void)in_sizes; (void)n_in; (void)out_size; (void)ws_size;
    const float* x  = (const float*)d_in[0];   // [B,S,D]
    const float* Wr = (const float*)d_in[1];   // [E,D]
    const float* br = (const float*)d_in[2];   // [E]
    const float* W1 = (const float*)d_in[3];   // [E,D,H]
    const float* b1 = (const float*)d_in[4];   // [E,H]
    const float* W2 = (const float*)d_in[5];   // [E,H,D]
    const float* b2 = (const float*)d_in[6];   // [E,D]
    float* y = (float*)d_out;                  // [B,S,D]

    // Workspace layout (all offsets 256B-aligned); total ~16.5 MB
    uint8_t* ws = (uint8_t*)d_ws;
    int*    cnt       = (int*)(ws);                                   // 8 ints
    int*    tok_list  = (int*)(ws + 256);                             // E*N ints   (256 KB)
    float*  gate_list = (float*)(ws + 256 + 262144);                  // E*N floats (256 KB)
    __bf16* W1t       = (__bf16*)(ws + 256 + 524288);                 // E*H*D bf16 (8 MB)
    __bf16* W2t       = (__bf16*)(ws + 256 + 524288 + 8388608);       // E*D*H bf16 (8 MB)

    hipMemsetAsync(d_out, 0, (size_t)N_TOK * D_DIM * sizeof(float), stream);
    hipMemsetAsync(cnt, 0, 256, stream);

    convert_w1_kernel<<<(E_NUM * H_DIM * D_DIM) / 256, 256, 0, stream>>>(W1, W1t);
    convert_w2_kernel<<<(E_NUM * D_DIM * H_DIM) / 256, 256, 0, stream>>>(W2, W2t);

    router_kernel<<<N_TOK / 8, 256, 0, stream>>>(x, Wr, br, cnt, tok_list, gate_list);

    ffn_kernel<<<E_NUM * MAX_TILES, 256, 0, stream>>>(x, W1t, b1, W2t, b2,
                                                      cnt, tok_list, gate_list, y);
}
